// AttModel_74045236183046
// MI455X (gfx1250) — matrix operand order repaired and checked by reading, not compiled, not run on hardware
//
#include <hip/hip_runtime.h>
#include <math.h>

// ---------------------------------------------------------------------------
// MI455X (gfx1250) fused attention:
//   v = relu(xWv+bv); q = relu(xWq+bq)*SCALE; k = relu(xWk+bk)
//   out = relu(softmax(q k^T masked) v)
// Flash-style online softmax, all matmuls via v_wmma_f32_16x16x32_f16.
// ---------------------------------------------------------------------------

typedef _Float16 f16;
typedef __attribute__((ext_vector_type(16))) _Float16 v16h;
typedef __attribute__((ext_vector_type(8)))  _Float16 v8h;
typedef __attribute__((ext_vector_type(8)))  float    v8f;

#define DIN   256
#define HID   512
#define NTOK  1024
#define NBAT  32

static constexpr float SCALE_F = 1.0f / 2304.0f;   // 1/(64*36)
static constexpr float NEGBIG  = 9e15f;

// Build a 16-bit A-matrix fragment (16x32) from a row pointer.
// A layout (ISA 7.12.2): lane m=lane%16; lanes<16 hold K = 0..7 (elts 0..7)
// and K = 16..23 (elts 8..15); lanes>=16 hold K = 8..15 / 24..31.
// Caller passes rowbase = &row[kstep*32 + 8*(lane>>4)].
__device__ __forceinline__ v16h load_a_frag(const f16* rowbase) {
    union { v16h v; v8h h[2]; } u;
    u.h[0] = *(const v8h*)(rowbase);
    u.h[1] = *(const v8h*)(rowbase + 16);
    return u.v;
}

// Build a 16-bit B-matrix fragment (32x16). B layout: lane n=lane%16;
// lanes 0-15 hold K=0..7 (elts 0..7) and K=16..23 (elts 8..15); lanes 16-31 hold K=8..15 / 24..31.
// Caller passes p = &col_row[k0 + 8*(lane>>4)] with K contiguous in memory.
__device__ __forceinline__ v16h load_b_frag(const f16* p) {
    union { v16h v; v8h h[2]; } u;
    u.h[0] = *(const v8h*)(p);
    u.h[1] = *(const v8h*)(p + 16);
    return u.v;
}

// ---------------------------------------------------------------------------
// Kernel 0: convert W [256x512] f32 -> Wt [512x256] f16 (transposed) so the
// projection B-fragments read two contiguous 16-byte runs per lane.
// ---------------------------------------------------------------------------
__global__ void wcvt_kernel(const float* __restrict__ Wq,
                            const float* __restrict__ Wk,
                            const float* __restrict__ Wv,
                            f16* __restrict__ Wt) {
    const int h     = blockIdx.x;      // 0..511
    const int which = blockIdx.y;      // 0=q 1=k 2=v
    const int k     = threadIdx.x;     // 0..255
    const float* W = (which == 0) ? Wq : (which == 1) ? Wk : Wv;
    Wt[((size_t)which * HID + h) * DIN + k] = (f16)W[(size_t)k * HID + h];
}

// ---------------------------------------------------------------------------
// Kernel 1: projection  out[16 x 512] = relu(x[16 x 256] @ W + b) (*scale)
// 4 waves per block; wave w owns H columns [128w, 128w+128).
// out_rm != null -> row-major f16 store (q/k); else transposed store (vT).
// ---------------------------------------------------------------------------
#define XPITCH (DIN + 8)   // pad LDS rows: 264 halves -> rows 4 banks apart

__global__ __launch_bounds__(128) void proj_kernel(
    const float* __restrict__ x,     // [B*N, 256] f32
    const f16*   __restrict__ Wt,    // [512, 256] f16 (transposed weights)
    const float* __restrict__ bias,  // [512]
    f16* __restrict__ out_rm,        // [B*N, 512] or null
    f16* __restrict__ out_tr,        // [B, 512, 1024] or null
    float scale) {
    __shared__ f16 Xs[16][XPITCH];

    const int tid  = threadIdx.x;
    const int lane = tid & 31;
    const int wave = tid >> 5;
    const int r0   = blockIdx.x * 16;            // row tile in [0, 32768)

    // cooperative x tile load + f32->f16 convert (tile is contiguous)
    for (int i = tid * 4; i < 16 * DIN; i += 128 * 4) {
        const float4 xv = *(const float4*)(x + (size_t)r0 * DIN + i);
        const int m = i >> 8, k = i & (DIN - 1);
        Xs[m][k + 0] = (f16)xv.x;
        Xs[m][k + 1] = (f16)xv.y;
        Xs[m][k + 2] = (f16)xv.z;
        Xs[m][k + 3] = (f16)xv.w;
    }
    __syncthreads();

    const int mA    = lane & 15;
    const int koffA = (lane >> 4) * 8;
    const int nB    = lane & 15;
    const int kbB   = (lane >> 4) * 8;
    const int hi    = lane >> 4;
    const int hw0   = wave * 128;

    v8f acc[8];
#pragma unroll
    for (int t = 0; t < 8; ++t)
#pragma unroll
        for (int e = 0; e < 8; ++e) acc[t][e] = 0.0f;

#pragma unroll
    for (int ks = 0; ks < 8; ++ks) {             // K = 256 in steps of 32
        const int k0 = ks * 32;
        const v16h a = load_a_frag(&Xs[mA][k0 + koffA]);
#pragma unroll
        for (int t = 0; t < 8; ++t) {
            const int h = hw0 + t * 16 + nB;
            const v16h b = load_b_frag(&Wt[(size_t)h * DIN + k0 + kbB]);
            acc[t] = __builtin_amdgcn_wmma_f32_16x16x32_f16(
                false, a, false, b, (short)0, acc[t], false, false);
        }
    }

    // epilogue: bias + relu (+scale), store f16
#pragma unroll
    for (int t = 0; t < 8; ++t) {
        const int h  = hw0 + t * 16 + nB;
        const float bv = bias[h];
#pragma unroll
        for (int i = 0; i < 8; ++i) {
            float v = fmaxf(acc[t][i] + bv, 0.0f) * scale;
            const int m  = i + 8 * hi;           // C layout row
            const int rg = r0 + m;
            if (out_rm) {
                out_rm[(size_t)rg * HID + h] = (f16)v;
            } else {
                const int b = rg >> 10, ml = rg & (NTOK - 1);
                out_tr[((size_t)b * HID + h) * NTOK + ml] = (f16)v;
            }
        }
    }
}

// ---------------------------------------------------------------------------
// Kernel 2: flash attention over one (batch, 16-row) tile per block.
// 4 waves: per 64-column step each wave computes one 16x16 S tile, then all
// waves share stats via LDS and accumulate their own 128-wide O chunk.
// ---------------------------------------------------------------------------
#define QPITCH (HID + 8)   // 520 halves -> rows 4 banks apart
#define SPITCH 68          // 68 floats  -> rows 4 banks apart

__global__ __launch_bounds__(128) void attn_kernel(
    const f16*   __restrict__ q16,   // [B, N, 512] f16, pre-scaled
    const f16*   __restrict__ k16,   // [B, N, 512] f16
    const f16*   __restrict__ vT,    // [B, 512, N] f16 (transposed)
    const float* __restrict__ mask,  // [B, N, N] f32
    float*       __restrict__ out) { // [B, N, 512] f32
    __shared__ f16   Qs[16][QPITCH];
    __shared__ float Swork[16][SPITCH];
    __shared__ float wmaxs[4][16];

    const int tid  = threadIdx.x;
    const int lane = tid & 31;
    const int wave = tid >> 5;
    const int b    = blockIdx.y;
    const int r0   = blockIdx.x * 16;

    // Q tile -> LDS (rows padded)
    {
        const f16* qb = q16 + ((size_t)b * NTOK + r0) * HID;
        for (int i = tid * 8; i < 16 * HID; i += 128 * 8) {
            const int m = i >> 9, k = i & (HID - 1);
            *(v8h*)(&Qs[m][k]) = *(const v8h*)(qb + i);
        }
    }
    __syncthreads();

    const int mA    = lane & 15;
    const int koffA = (lane >> 4) * 8;
    const int nB    = lane & 15;
    const int kbB   = (lane >> 4) * 8;
    const int hi    = lane >> 4;
    const int hw0   = wave * 128;
    const int cw    = wave * 16;

    const f16*   kbat = k16  + (size_t)b * NTOK * HID;
    const f16*   vbat = vT   + (size_t)b * HID * NTOK;
    const float* mbat = mask + (size_t)b * NTOK * NTOK;

    v8f acc[8];
#pragma unroll
    for (int t = 0; t < 8; ++t)
#pragma unroll
        for (int e = 0; e < 8; ++e) acc[t][e] = 0.0f;

    float m_run = -INFINITY;
    float l_run = 0.0f;

    for (int it = 0; it < NTOK / 64; ++it) {
        const int c0 = it * 64;

        // ---- phase 1: S[16x16] = Q (K chunk)^T, K-dim = 512 ----
        v8f s;
#pragma unroll
        for (int e = 0; e < 8; ++e) s[e] = 0.0f;

        const f16* krow = kbat + (size_t)(c0 + cw + nB) * HID + kbB;
        if (it + 1 < NTOK / 64)   // gfx1250 global_prefetch_b8 for next K tile
            __builtin_prefetch(krow + 64 * HID, 0, 0);
#pragma unroll
        for (int ks = 0; ks < 16; ++ks) {
            const v16h a  = load_a_frag(&Qs[mA][ks * 32 + koffA]);
            const v16h bf = load_b_frag(krow + ks * 32);
            s = __builtin_amdgcn_wmma_f32_16x16x32_f16(
                false, a, false, bf, (short)0, s, false, false);
        }

        // mask, stash S strip, local row max
#pragma unroll
        for (int i = 0; i < 8; ++i) {
            const int m  = i + 8 * hi;
            const float mk = mbat[(size_t)(r0 + m) * NTOK + c0 + cw + nB];
            const float sv = s[i] * mk - NEGBIG * (1.0f - mk);
            s[i] = sv;
            Swork[m][cw + nB] = sv;
        }
#pragma unroll
        for (int i = 0; i < 8; ++i) {
            float v = s[i];
            v = fmaxf(v, __shfl_xor(v, 1));
            v = fmaxf(v, __shfl_xor(v, 2));
            v = fmaxf(v, __shfl_xor(v, 4));
            v = fmaxf(v, __shfl_xor(v, 8));
            if ((lane & 15) == 0) wmaxs[wave][i + 8 * hi] = v;
        }
        __syncthreads();

        // ---- phase 2: shared flash stats (each wave recomputes, identical) ----
        const float mloc = fmaxf(fmaxf(wmaxs[0][mA], wmaxs[1][mA]),
                                 fmaxf(wmaxs[2][mA], wmaxs[3][mA]));
        const float m_new = fmaxf(m_run, mloc);
        const float alpha = __expf(m_run - m_new);

        // P = exp(S - m_new) as two A fragments (cols 0..31, 32..63) + rowsum
        float rsum = 0.0f;
        v16h pa[2];
#pragma unroll
        for (int t = 0; t < 2; ++t) {
            union { v16h v; v8h h[2]; } u;
#pragma unroll
            for (int j = 0; j < 8; ++j) {
                const float p0 = __expf(Swork[mA][t * 32 + koffA + j]      - m_new);
                const float p1 = __expf(Swork[mA][t * 32 + koffA + 16 + j] - m_new);
                rsum += p0 + p1;
                u.h[0][j] = (f16)p0;
                u.h[1][j] = (f16)p1;
            }
            pa[t] = u.v;
        }
        rsum += __shfl_xor(rsum, 16);       // lane^16 holds the other K half
        l_run = l_run * alpha + rsum;
        m_run = m_new;

        // rescale O accumulator (alpha indexed by C-layout row via shuffle)
#pragma unroll
        for (int i = 0; i < 8; ++i) {
            const float ac = __shfl(alpha, i + 8 * hi);
#pragma unroll
            for (int t = 0; t < 8; ++t) acc[t][i] *= ac;
        }

        // O[16x128] += P[16x64] @ V[64x128], V read from transposed vT
#pragma unroll
        for (int t2 = 0; t2 < 2; ++t2) {
#pragma unroll
            for (int hs = 0; hs < 8; ++hs) {
                const f16* vp = vbat + (size_t)(hw0 + hs * 16 + nB) * NTOK
                                     + c0 + t2 * 32 + kbB;
                const v16h bf = load_b_frag(vp);
                acc[hs] = __builtin_amdgcn_wmma_f32_16x16x32_f16(
                    false, pa[t2], false, bf, (short)0, acc[hs], false, false);
            }
        }
        __syncthreads();   // Swork/wmaxs reused next iteration
    }

    // ---- epilogue: out = relu(O / l) ----
#pragma unroll
    for (int i = 0; i < 8; ++i) {
        const float lc  = __shfl(l_run, i + 8 * hi);
        const float inv = 1.0f / lc;
        const int   m   = i + 8 * hi;
#pragma unroll
        for (int hs = 0; hs < 8; ++hs) {
            const float v = fmaxf(acc[hs][i] * inv, 0.0f);
            out[((size_t)b * NTOK + r0 + m) * HID + hw0 + hs * 16 + nB] = v;
        }
    }
}

// ---------------------------------------------------------------------------
// Launcher. Input order: x, mask, Wv, bv, Wk, bk, Wq, bq.
// Workspace: Wt(3*512*256 f16) | q16 | k16 | vT (each 32*1024*512 f16).
// ---------------------------------------------------------------------------
extern "C" void kernel_launch(void* const* d_in, const int* in_sizes, int n_in,
                              void* d_out, int out_size, void* d_ws, size_t ws_size,
                              hipStream_t stream) {
    const float* x    = (const float*)d_in[0];
    const float* mask = (const float*)d_in[1];
    const float* Wv   = (const float*)d_in[2];
    const float* bv   = (const float*)d_in[3];
    const float* Wk   = (const float*)d_in[4];
    const float* bk   = (const float*)d_in[5];
    const float* Wq   = (const float*)d_in[6];
    const float* bq   = (const float*)d_in[7];
    float* out = (float*)d_out;

    char* ws = (char*)d_ws;
    const size_t WT_BYTES  = (size_t)3 * HID * DIN * sizeof(f16);      // 768 KB
    const size_t QKV_BYTES = (size_t)NBAT * NTOK * HID * sizeof(f16);  // 32 MB
    f16* Wt  = (f16*)ws;
    f16* q16 = (f16*)(ws + WT_BYTES);
    f16* k16 = (f16*)(ws + WT_BYTES + QKV_BYTES);
    f16* vT  = (f16*)(ws + WT_BYTES + 2 * QKV_BYTES);

    wcvt_kernel<<<dim3(HID, 3), DIN, 0, stream>>>(Wq, Wk, Wv, Wt);

    const int nRowTiles = NBAT * NTOK / 16;   // 2048
    proj_kernel<<<dim3(nRowTiles), 128, 0, stream>>>(
        x, Wt + (size_t)0 * HID * DIN, bq, q16, nullptr, SCALE_F);
    proj_kernel<<<dim3(nRowTiles), 128, 0, stream>>>(
        x, Wt + (size_t)1 * HID * DIN, bk, k16, nullptr, 1.0f);
    proj_kernel<<<dim3(nRowTiles), 128, 0, stream>>>(
        x, Wt + (size_t)2 * HID * DIN, bv, nullptr, vT, 1.0f);

    attn_kernel<<<dim3(NTOK / 16, NBAT), 128, 0, stream>>>(
        q16, k16, vT, mask, out);
}
